// MultiHeadAttention_13013750907532
// MI455X (gfx1250) — matrix operand order repaired and checked by reading, not compiled, not run on hardware
//
#include <hip/hip_runtime.h>
#include <hip/hip_bf16.h>

// ---- problem constants (fixed by reference) ----
#define BB   4
#define TQn  1024
#define TKVn 1024
#define Dn   1024
#define NH   16
#define HDn  64   // head size

typedef __attribute__((ext_vector_type(16))) __bf16 v16bf;
typedef __attribute__((ext_vector_type(8)))  float  v8f;

union Frag16 { v16bf v; unsigned u[8]; unsigned short h[16]; };
union FragF  { v8f   v; float f[8]; };

__device__ __forceinline__ unsigned short f2bf(float f) {
    union { float f; unsigned u; } x; x.f = f;
    unsigned r = x.u + 0x7fffu + ((x.u >> 16) & 1u);
    return (unsigned short)(r >> 16);
}

// A-fragment (16-bit, 16x32) K base for dword d, contiguous order (k = 16*hi + 2*d).
__device__ __forceinline__ int a_kbase(int d, int hi) {
    return 2 * d + hi * 16;
}

// ---------------- elementwise converts ----------------
__global__ void cast_bf16_k(const float* __restrict__ src,
                            unsigned short* __restrict__ dst, int n) {
    int i = blockIdx.x * blockDim.x + threadIdx.x;
    if (i < n) dst[i] = f2bf(src[i]);
}

// Wt[n*K + k] = bf16(W[k*N + n])
__global__ void transpose_bf16_k(const float* __restrict__ W,
                                 unsigned short* __restrict__ Wt, int K, int N) {
    int i = blockIdx.x * blockDim.x + threadIdx.x;
    if (i >= K * N) return;
    int k = i / N, n = i - k * N;
    Wt[(size_t)n * K + k] = f2bf(W[i]);
}

// ---------------- fragment load helpers (all indices compile-time) ----------
__device__ __forceinline__ void load_afrag(Frag16& a, const unsigned* __restrict__ arow,
                                           int k0, int hi) {
#pragma unroll
    for (int d = 0; d < 8; ++d) a.u[d] = arow[(k0 + a_kbase(d, hi)) >> 1];
}

__device__ __forceinline__ void load_bfrags8(Frag16* b, const unsigned* const* brow,
                                             int k0, int hi) {
#pragma unroll
    for (int j = 0; j < 8; ++j)
#pragma unroll
        for (int d = 0; d < 8; ++d)
            b[j].u[d] = brow[j][(k0 + 2 * d + hi * 16) >> 1];
}

// ---------------- generic WMMA GEMM: C(MxN,f32) = A(MxK,bf16) * Wt(NxK,bf16)^T
// One wave computes a 16x128 tile. The k-loop is manually unrolled 2x with two
// statically-named fragment buffers so loads of step k+32 overlap the WMMAs of
// step k without any dynamically-indexed (spill-prone) arrays.
// Requires K % 64 == 0 (true here: K = 1024 everywhere).
__global__ void __launch_bounds__(32)
gemm_bf16_tn(const unsigned short* __restrict__ A,
             const unsigned short* __restrict__ Wt,
             float* __restrict__ C, int M, int K, int N) {
    const int l  = threadIdx.x;
    const int hi = (l >> 4) & 1;
    const int ln = l & 15;
    const int m0 = blockIdx.x * 16;
    const int n0 = blockIdx.y * 128;

    FragF acc[8];
#pragma unroll
    for (int j = 0; j < 8; ++j)
#pragma unroll
        for (int i = 0; i < 8; ++i) acc[j].f[i] = 0.0f;

    const unsigned* arow = (const unsigned*)(A + (size_t)(m0 + ln) * K);
    const unsigned* brow[8];
#pragma unroll
    for (int j = 0; j < 8; ++j)
        brow[j] = (const unsigned*)(Wt + (size_t)(n0 + j * 16 + ln) * K);

    Frag16 a0, a1;
    Frag16 b0[8], b1[8];

    load_afrag(a0, arow, 0, hi);
    load_bfrags8(b0, brow, 0, hi);

    for (int k0 = 0; k0 < K; k0 += 64) {
        // stage k0+32 while computing on k0
        load_afrag(a1, arow, k0 + 32, hi);
        load_bfrags8(b1, brow, k0 + 32, hi);
        __builtin_prefetch((const char*)arow + ((size_t)(k0 + 64) << 1), 0, 3);
#pragma unroll
        for (int j = 0; j < 8; ++j)
            acc[j].v = __builtin_amdgcn_wmma_f32_16x16x32_bf16(
                false, a0.v, false, b0[j].v, (short)0, acc[j].v, false, false);
        // stage k0+64 while computing on k0+32
        if (k0 + 64 < K) {
            load_afrag(a0, arow, k0 + 64, hi);
            load_bfrags8(b0, brow, k0 + 64, hi);
        }
#pragma unroll
        for (int j = 0; j < 8; ++j)
            acc[j].v = __builtin_amdgcn_wmma_f32_16x16x32_bf16(
                false, a1.v, false, b1[j].v, (short)0, acc[j].v, false, false);
    }

#pragma unroll
    for (int j = 0; j < 8; ++j)
#pragma unroll
        for (int i = 0; i < 8; ++i)
            C[(size_t)(m0 + i + hi * 8) * N + n0 + j * 16 + ln] = acc[j].f[i];
}

// ---------------- head split + RMSNorm ----------------
// qproj: (B*TQ) x 2D.  q1h/q2h: (B,H,TQ,HD) bf16
__global__ void qheads_k(const float* __restrict__ qproj,
                         const float* __restrict__ qn1, const float* __restrict__ qn2,
                         unsigned short* __restrict__ q1h,
                         unsigned short* __restrict__ q2h) {
    int gid = blockIdx.x * blockDim.x + threadIdx.x;
    if (gid >= BB * TQn * NH) return;
    int h = gid % NH, bt = gid / NH, t = bt % TQn, b = bt / TQn;
    const float* r = qproj + (size_t)bt * (2 * Dn) + h * HDn;
    float s1 = 0.f, s2 = 0.f;
    for (int d = 0; d < HDn; ++d) { float a = r[d]; s1 += a * a; float c = r[Dn + d]; s2 += c * c; }
    float r1 = rsqrtf(s1 / HDn + 1e-5f), r2 = rsqrtf(s2 / HDn + 1e-5f);
    size_t o = ((size_t)(b * NH + h) * TQn + t) * HDn;
    for (int d = 0; d < HDn; ++d) {
        q1h[o + d] = f2bf(r[d] * r1 * qn1[d]);
        q2h[o + d] = f2bf(r[Dn + d] * r2 * qn2[d]);
    }
}

// kvproj: (B*TKV) x 3D. k1h/k2h: (B,H,TKV,HD); vt: (B,H,HD,TKV) (transposed!)
__global__ void kvheads_k(const float* __restrict__ kvproj,
                          const float* __restrict__ kn1, const float* __restrict__ kn2,
                          unsigned short* __restrict__ k1h,
                          unsigned short* __restrict__ k2h,
                          unsigned short* __restrict__ vt) {
    int gid = blockIdx.x * blockDim.x + threadIdx.x;
    if (gid >= BB * TKVn * NH) return;
    int h = gid % NH, bt = gid / NH, t = bt % TKVn, b = bt / TKVn;
    const float* r = kvproj + (size_t)bt * (3 * Dn) + h * HDn;
    float s1 = 0.f, s2 = 0.f;
    for (int d = 0; d < HDn; ++d) { float a = r[d]; s1 += a * a; float c = r[Dn + d]; s2 += c * c; }
    float r1 = rsqrtf(s1 / HDn + 1e-5f), r2 = rsqrtf(s2 / HDn + 1e-5f);
    size_t o = ((size_t)(b * NH + h) * TKVn + t) * HDn;
    size_t vbase = (size_t)(b * NH + h) * HDn * TKVn;
    for (int d = 0; d < HDn; ++d) {
        k1h[o + d] = f2bf(r[d] * r1 * kn1[d]);
        k2h[o + d] = f2bf(r[Dn + d] * r2 * kn2[d]);
        vt[vbase + (size_t)d * TKVn + t] = f2bf(r[2 * Dn + d]);
    }
}

// ---------------- differential flash attention ----------------
// One wave handles 16 queries of one (b,h). y: (B,H,TQ,HD) f32.
__global__ void __launch_bounds__(32)
diff_attn_k(const unsigned short* __restrict__ q1h,
            const unsigned short* __restrict__ q2h,
            const unsigned short* __restrict__ k1h,
            const unsigned short* __restrict__ k2h,
            const unsigned short* __restrict__ vt,
            const float* __restrict__ lmb,
            float* __restrict__ y) {
    __shared__ unsigned short ldsP1[16 * 32];
    __shared__ unsigned short ldsP2[16 * 32];

    const int l = threadIdx.x, hi = (l >> 4) & 1, ln = l & 15;
    const int qt = blockIdx.x;        // TQ/16 tiles
    const int bh = blockIdx.y;        // B*H
    const int h  = bh & (NH - 1);
    const float lam = log1pf(__expf(lmb[h]));   // softplus

    const size_t baseQ = (size_t)bh * TQn * HDn + (size_t)qt * 16 * HDn;
    const size_t baseK = (size_t)bh * TKVn * HDn;
    const size_t baseV = (size_t)bh * HDn * TKVn;

    // Q fragments, two K-chunks (head-dim 0..31, 32..63) per branch
    Frag16 q1f[2], q2f[2];
    {
        const unsigned* q1row = (const unsigned*)(q1h + baseQ + (size_t)ln * HDn);
        const unsigned* q2row = (const unsigned*)(q2h + baseQ + (size_t)ln * HDn);
#pragma unroll
        for (int c = 0; c < 2; ++c)
#pragma unroll
            for (int d = 0; d < 8; ++d) {
                int kb = c * 32 + a_kbase(d, hi);
                q1f[c].u[d] = q1row[kb >> 1];
                q2f[c].u[d] = q2row[kb >> 1];
            }
    }

    float m1[8], s1[8], m2[8], s2[8];
    FragF o1[4], o2[4];
#pragma unroll
    for (int i = 0; i < 8; ++i) { m1[i] = m2[i] = -1e30f; s1[i] = s2[i] = 0.f; }
#pragma unroll
    for (int j = 0; j < 4; ++j)
#pragma unroll
        for (int i = 0; i < 8; ++i) { o1[j].f[i] = 0.f; o2[j].f[i] = 0.f; }

    const float scale = 0.125f;  // 1/sqrt(64)

    for (int kt = 0; kt < TKVn; kt += 32) {
        // ---- S = Q K^T for both branches, 2 key-subtiles of 16 ----
        FragF sa1, sb1, sa2, sb2;
#pragma unroll
        for (int sub = 0; sub < 2; ++sub) {
            Frag16 kb1[2], kb2[2];
            const unsigned* k1row = (const unsigned*)(k1h + baseK + (size_t)(kt + sub * 16 + ln) * HDn);
            const unsigned* k2row = (const unsigned*)(k2h + baseK + (size_t)(kt + sub * 16 + ln) * HDn);
#pragma unroll
            for (int c = 0; c < 2; ++c)
#pragma unroll
                for (int d = 0; d < 8; ++d) {
                    int off = (c * 32 + 2 * d + hi * 16) >> 1;
                    kb1[c].u[d] = k1row[off];
                    kb2[c].u[d] = k2row[off];
                }
            FragF a1, a2;
#pragma unroll
            for (int i = 0; i < 8; ++i) { a1.f[i] = 0.f; a2.f[i] = 0.f; }
            a1.v = __builtin_amdgcn_wmma_f32_16x16x32_bf16(false, q1f[0].v, false, kb1[0].v, (short)0, a1.v, false, false);
            a1.v = __builtin_amdgcn_wmma_f32_16x16x32_bf16(false, q1f[1].v, false, kb1[1].v, (short)0, a1.v, false, false);
            a2.v = __builtin_amdgcn_wmma_f32_16x16x32_bf16(false, q2f[0].v, false, kb2[0].v, (short)0, a2.v, false, false);
            a2.v = __builtin_amdgcn_wmma_f32_16x16x32_bf16(false, q2f[1].v, false, kb2[1].v, (short)0, a2.v, false, false);
            if (sub == 0) { sa1 = a1; sa2 = a2; } else { sb1 = a1; sb2 = a2; }
        }

        __syncthreads();  // protect LDS against previous iteration's readers

        // ---- online softmax (row r = i + 8*hi lives in one 16-lane half) ----
#pragma unroll
        for (int i = 0; i < 8; ++i) {
            // branch 1
            float va = sa1.f[i] * scale, vb = sb1.f[i] * scale;
            float t = fmaxf(va, vb);
#pragma unroll
            for (int mm = 1; mm < 16; mm <<= 1) t = fmaxf(t, __shfl_xor(t, mm, 16));
            float mn = fmaxf(m1[i], t);
            float corr = __expf(m1[i] - mn);
            float pa = __expf(va - mn), pb = __expf(vb - mn);
            float rs = pa + pb;
#pragma unroll
            for (int mm = 1; mm < 16; mm <<= 1) rs += __shfl_xor(rs, mm, 16);
            s1[i] = s1[i] * corr + rs; m1[i] = mn;
#pragma unroll
            for (int j = 0; j < 4; ++j) o1[j].f[i] *= corr;
            ldsP1[(i + hi * 8) * 32 + ln]      = f2bf(pa);
            ldsP1[(i + hi * 8) * 32 + 16 + ln] = f2bf(pb);
            // branch 2
            va = sa2.f[i] * scale; vb = sb2.f[i] * scale;
            t = fmaxf(va, vb);
#pragma unroll
            for (int mm = 1; mm < 16; mm <<= 1) t = fmaxf(t, __shfl_xor(t, mm, 16));
            mn = fmaxf(m2[i], t);
            corr = __expf(m2[i] - mn);
            pa = __expf(va - mn); pb = __expf(vb - mn);
            rs = pa + pb;
#pragma unroll
            for (int mm = 1; mm < 16; mm <<= 1) rs += __shfl_xor(rs, mm, 16);
            s2[i] = s2[i] * corr + rs; m2[i] = mn;
#pragma unroll
            for (int j = 0; j < 4; ++j) o2[j].f[i] *= corr;
            ldsP2[(i + hi * 8) * 32 + ln]      = f2bf(pa);
            ldsP2[(i + hi * 8) * 32 + 16 + ln] = f2bf(pb);
        }
        __syncthreads();

        // ---- O += P * V ----
        Frag16 p1, p2;
        {
            const unsigned* p1row = (const unsigned*)(ldsP1 + ln * 32);
            const unsigned* p2row = (const unsigned*)(ldsP2 + ln * 32);
#pragma unroll
            for (int d = 0; d < 8; ++d) {
                int kb = a_kbase(d, hi);
                p1.u[d] = p1row[kb >> 1];
                p2.u[d] = p2row[kb >> 1];
            }
        }
#pragma unroll
        for (int j = 0; j < 4; ++j) {
            Frag16 vb;
            const unsigned* vrow =
                (const unsigned*)(vt + baseV + (size_t)(j * 16 + ln) * TKVn + kt);
#pragma unroll
            for (int d = 0; d < 8; ++d) vb.u[d] = vrow[d + hi * 8];
            o1[j].v = __builtin_amdgcn_wmma_f32_16x16x32_bf16(false, p1.v, false, vb.v, (short)0, o1[j].v, false, false);
            o2[j].v = __builtin_amdgcn_wmma_f32_16x16x32_bf16(false, p2.v, false, vb.v, (short)0, o2[j].v, false, false);
        }
    }

    // ---- epilogue: y = O1/l1 - lam * O2/l2 ----
#pragma unroll
    for (int j = 0; j < 4; ++j)
#pragma unroll
        for (int i = 0; i < 8; ++i) {
            float r1 = o1[j].f[i] / s1[i];
            float r2 = o2[j].f[i] / s2[i];
            y[baseQ + (size_t)(i + hi * 8) * HDn + j * 16 + ln] = r1 - lam * r2;
        }
}

// ---------------- GroupNorm over (TQ,HD) per (b,h), emits (B,TQ,D) bf16 ----
__global__ void groupnorm_k(const float* __restrict__ y,
                            const float* __restrict__ gw, const float* __restrict__ gb,
                            unsigned short* __restrict__ yn) {
    __shared__ float ssum[256], ssq[256];
    int bh = blockIdx.x, b = bh / NH, h = bh % NH;
    const float* yp = y + (size_t)bh * TQn * HDn;
    float s = 0.f, q = 0.f;
    for (int i = threadIdx.x; i < TQn * HDn; i += 256) { float v = yp[i]; s += v; q += v * v; }
    ssum[threadIdx.x] = s; ssq[threadIdx.x] = q; __syncthreads();
    for (int st = 128; st > 0; st >>= 1) {
        if ((int)threadIdx.x < st) { ssum[threadIdx.x] += ssum[threadIdx.x + st]; ssq[threadIdx.x] += ssq[threadIdx.x + st]; }
        __syncthreads();
    }
    float inv = 1.0f / (TQn * HDn);
    float mu = ssum[0] * inv;
    float var = ssq[0] * inv - mu * mu;
    float rstd = rsqrtf(var + 1e-5f);
    for (int i = threadIdx.x; i < TQn * HDn; i += 256) {
        int t = i / HDn, d = i - t * HDn;
        float v = (yp[i] - mu) * rstd * gw[h * HDn + d] + gb[h * HDn + d];
        yn[(size_t)b * TQn * Dn + (size_t)t * Dn + h * HDn + d] = f2bf(v);
    }
}

// ---------------- driver ----------------
extern "C" void kernel_launch(void* const* d_in, const int* in_sizes, int n_in,
                              void* d_out, int out_size, void* d_ws, size_t ws_size,
                              hipStream_t stream) {
    (void)in_sizes; (void)n_in; (void)out_size; (void)ws_size;
    const float* x_q  = (const float*)d_in[0];
    const float* x_kv = (const float*)d_in[1];
    const float* Wq   = (const float*)d_in[2];
    const float* Wkv  = (const float*)d_in[3];
    const float* Wc   = (const float*)d_in[4];
    const float* qn1  = (const float*)d_in[5];
    const float* kn1  = (const float*)d_in[6];
    const float* qn2  = (const float*)d_in[7];
    const float* kn2  = (const float*)d_in[8];
    const float* gn_w = (const float*)d_in[9];
    const float* gn_b = (const float*)d_in[10];
    const float* lmb  = (const float*)d_in[11];
    float* out = (float*)d_out;

    char* ws = (char*)d_ws; size_t off = 0;
    auto alloc = [&](size_t bytes) -> void* {
        void* p = ws + off; off += (bytes + 255) & ~(size_t)255; return p;
    };

    const size_t NTOK = (size_t)BB * TQn;            // 4096 rows
    const size_t NEH  = (size_t)BB * NH * TQn * HDn; // 4M head elems

    unsigned short* xq_bf  = (unsigned short*)alloc(NTOK * Dn * 2);
    unsigned short* xkv_bf = (unsigned short*)alloc(NTOK * Dn * 2);
    unsigned short* wq_t   = (unsigned short*)alloc((size_t)Dn * 2 * Dn * 2);
    unsigned short* wkv_t  = (unsigned short*)alloc((size_t)Dn * 3 * Dn * 2);
    unsigned short* wc_t   = (unsigned short*)alloc((size_t)Dn * Dn * 2);
    float* qproj  = (float*)alloc(NTOK * 2 * Dn * 4);
    float* kvproj = (float*)alloc(NTOK * 3 * Dn * 4);
    unsigned short* q1h = (unsigned short*)alloc(NEH * 2);
    unsigned short* q2h = (unsigned short*)alloc(NEH * 2);
    unsigned short* k1h = (unsigned short*)alloc(NEH * 2);
    unsigned short* k2h = (unsigned short*)alloc(NEH * 2);
    unsigned short* vt  = (unsigned short*)alloc(NEH * 2);
    float* ybuf = (float*)alloc(NEH * 4);
    unsigned short* yn = (unsigned short*)alloc(NEH * 2);

    int n;
    n = (int)(NTOK * Dn);
    cast_bf16_k<<<(n + 255) / 256, 256, 0, stream>>>(x_q, xq_bf, n);
    cast_bf16_k<<<(n + 255) / 256, 256, 0, stream>>>(x_kv, xkv_bf, n);
    n = Dn * 2 * Dn;
    transpose_bf16_k<<<(n + 255) / 256, 256, 0, stream>>>(Wq, wq_t, Dn, 2 * Dn);
    n = Dn * 3 * Dn;
    transpose_bf16_k<<<(n + 255) / 256, 256, 0, stream>>>(Wkv, wkv_t, Dn, 3 * Dn);
    n = Dn * Dn;
    transpose_bf16_k<<<(n + 255) / 256, 256, 0, stream>>>(Wc, wc_t, Dn, Dn);

    gemm_bf16_tn<<<dim3((int)NTOK / 16, (2 * Dn) / 128), 32, 0, stream>>>(
        xq_bf, wq_t, qproj, (int)NTOK, Dn, 2 * Dn);
    gemm_bf16_tn<<<dim3((int)NTOK / 16, (3 * Dn) / 128), 32, 0, stream>>>(
        xkv_bf, wkv_t, kvproj, (int)NTOK, Dn, 3 * Dn);

    n = BB * TQn * NH;
    qheads_k<<<(n + 127) / 128, 128, 0, stream>>>(qproj, qn1, qn2, q1h, q2h);
    n = BB * TKVn * NH;
    kvheads_k<<<(n + 127) / 128, 128, 0, stream>>>(kvproj, kn1, kn2, k1h, k2h, vt);

    diff_attn_k<<<dim3(TQn / 16, BB * NH), 32, 0, stream>>>(
        q1h, q2h, k1h, k2h, vt, lmb, ybuf);

    groupnorm_k<<<BB * NH, 256, 0, stream>>>(ybuf, gn_w, gn_b, yn);

    gemm_bf16_tn<<<dim3((int)NTOK / 16, Dn / 128), 32, 0, stream>>>(
        yn, wc_t, out, (int)NTOK, Dn, Dn);
}
